// PfaffianNet_18056042512517
// MI455X (gfx1250) — matrix operand order repaired and checked by reading, not compiled, not run on hardware
//
#include <hip/hip_runtime.h>

typedef _Float16 half8 __attribute__((ext_vector_type(8)));
typedef _Float16 v16h  __attribute__((ext_vector_type(16)));
typedef float    v8f   __attribute__((ext_vector_type(8)));

#define NPART 6
#define NBAS  64
#define BPW   8                 // batch elements per wave/block
#define ROWS  (NPART * BPW)     // 48 particle-rows
#define RT    3                 // 16-row WMMA tiles
#define PAIRS (BPW * 36)        // 288
#define PT    (PAIRS / 16)      // 18 MLP tiles

static __device__ __forceinline__ v16h cat8(half8 lo, half8 hi) {
  return __builtin_shufflevector(lo, hi, 0,1,2,3,4,5,6,7,8,9,10,11,12,13,14,15);
}
static __device__ __forceinline__ float silu_f(float z) {
  return z / (1.0f + expf(-z));
}

__global__ __launch_bounds__(32)
void pfaffian6_fused(const float* __restrict__ xg, const int* __restrict__ spg,
                     const float* __restrict__ Fud, const float* __restrict__ Fuu,
                     const float* __restrict__ Fdd, const float* __restrict__ W1g,
                     const float* __restrict__ b1g, const float* __restrict__ W2g,
                     const float* __restrict__ b2g, const float* __restrict__ W3g,
                     const float* __restrict__ b3g, const float* __restrict__ scg,
                     float* __restrict__ outp, int nB)
{
  // ---- LDS (≈54 KB/block) ----
  __shared__ __align__(16) _Float16 sFt[3][64][64];   // [s][n][k] = F_s[k][n]  (f16)
  __shared__ __align__(16) _Float16 sW2t[32][32];     // [n][k]    = W2[k][n]   (f16)
  __shared__ __align__(16) _Float16 sPhi[ROWS][64];   // Phi rows               (f16)
  __shared__ __align__(16) _Float16 sG[3][ROWS][64];  // G_s = Phi @ F_s        (f16)
  __shared__ float sW1[4][32];
  __shared__ float sb1[32], sb2[32], sW3[32];
  __shared__ float sX[ROWS][2];
  __shared__ int   sSpin[ROWS];
  __shared__ float sAp[PAIRS];                        // pairing A entries (spin-combined)
  __shared__ float sOut[PAIRS];                       // MLP out per (b,i,j)

  const int lane = threadIdx.x;
  const int h    = lane >> 4;          // wave half (WMMA K-split)
  const int l15  = lane & 15;
  const int r0   = blockIdx.x * ROWS;  // first global particle-row

  __builtin_prefetch(xg + (size_t)r0 * 2, 0, 1);   // global_prefetch_b8

  const float sc  = scg[0];
  const float b3v = b3g[0];

  // ---- Stage 0: params + positions into LDS ----
  for (int idx = lane; idx < 4096; idx += 32) {          // idx = k*64 + n
    int k = idx >> 6, n = idx & 63;
    sFt[0][n][k] = (_Float16)(Fuu[idx] - Fuu[n * 64 + k]);   // F_uu - F_uu^T
    sFt[1][n][k] = (_Float16)(Fdd[idx] - Fdd[n * 64 + k]);   // F_dd - F_dd^T
    sFt[2][n][k] = (_Float16)Fud[idx];
  }
  for (int idx = lane; idx < 1024; idx += 32)            // idx = k*32 + n
    sW2t[idx & 31][idx >> 5] = (_Float16)W2g[idx];
  for (int idx = lane; idx < 128; idx += 32)
    sW1[idx >> 5][idx & 31] = W1g[idx];
  sb1[lane] = b1g[lane];
  sb2[lane] = b2g[lane];
  sW3[lane] = W3g[lane];
  for (int idx = lane; idx < ROWS * 2; idx += 32)
    ((float*)sX)[idx] = xg[(size_t)r0 * 2 + idx];
  for (int idx = lane; idx < ROWS; idx += 32)
    sSpin[idx] = spg[r0 + idx];
  __syncthreads();

  // ---- Stage 1: Phi (Hermite-Gaussian) -> WMMA A-frags -> G_s tiles ----
  for (int t = 0; t < RT; ++t) {
    const int row = t * 16 + l15;        // both lane-halves: same row, different K halves
    const float u = sX[row][0], w = sX[row][1];

    float hx[8], hy[8];
    hx[0] = 1.0f; hx[1] = 2.0f * u;
    hy[0] = 1.0f; hy[1] = 2.0f * w;
#pragma unroll
    for (int k = 2; k < 8; ++k) {
      hx[k] = 2.0f * u * hx[k - 1] - 2.0f * (float)(k - 1) * hx[k - 2];
      hy[k] = 2.0f * w * hy[k - 1] - 2.0f * (float)(k - 1) * hy[k - 2];
    }
    const float g = expf(-0.5f * (u * u + w * w));
    float nm = 0.7511255444649425f;      // pi^{-1/4}; nm_m = nm_{m-1}/sqrt(2m)
    hx[0] *= nm * g;  hy[0] *= nm;       // fold gaussian into hx
#pragma unroll
    for (int m = 1; m < 8; ++m) {
      nm *= (1.0f / sqrtf(2.0f * (float)m));
      hx[m] *= nm * g;
      hy[m] *= nm;
    }

    // A-fragments (ISA A layout: half0 K in {0-7,16-23}, half1 {8-15,24-31})
    v16h a0, a1;
    {
      const float hxa0 = h ? hx[1] : hx[0];   // frag0: nx = h, 2+h
      const float hxb0 = h ? hx[3] : hx[2];
      const float hxa1 = h ? hx[5] : hx[4];   // frag1: nx = 4+h, 6+h
      const float hxb1 = h ? hx[7] : hx[6];
#pragma unroll
      for (int e = 0; e < 8; ++e) {
        a0[e]     = (_Float16)(hxa0 * hy[e]);
        a0[e + 8] = (_Float16)(hxb0 * hy[e]);
        a1[e]     = (_Float16)(hxa1 * hy[e]);
        a1[e + 8] = (_Float16)(hxb1 * hy[e]);
      }
    }
    // scatter the same halves into LDS -> full f16 Phi rows for the contraction
    *(half8*)&sPhi[row][     h * 8] = __builtin_shufflevector(a0, a0, 0,1,2,3,4,5,6,7);
    *(half8*)&sPhi[row][16 + h * 8] = __builtin_shufflevector(a0, a0, 8,9,10,11,12,13,14,15);
    *(half8*)&sPhi[row][32 + h * 8] = __builtin_shufflevector(a1, a1, 0,1,2,3,4,5,6,7);
    *(half8*)&sPhi[row][48 + h * 8] = __builtin_shufflevector(a1, a1, 8,9,10,11,12,13,14,15);

    // G_s[row-tile] = Phi_tile @ F_s : 3 F x 4 N-tiles x 2 K-frags = 24 WMMAs
#pragma unroll
    for (int s = 0; s < 3; ++s) {
#pragma unroll
      for (int nt = 0; nt < 4; ++nt) {
        const int ncol = nt * 16 + l15;
        const _Float16* fb = &sFt[s][ncol][0];
        // B layout: half0 K in {0-7,16-23}, half1 {8-15,24-31} (matches A)
        v16h b0 = cat8(*(const half8*)(fb +      h * 8), *(const half8*)(fb + 16 + h * 8));
        v16h b1 = cat8(*(const half8*)(fb + 32 + h * 8), *(const half8*)(fb + 48 + h * 8));
        v8f acc = {0.f,0.f,0.f,0.f,0.f,0.f,0.f,0.f};
        acc = __builtin_amdgcn_wmma_f32_16x16x32_f16(false, a0, false, b0, (short)0, acc, false, false);
        acc = __builtin_amdgcn_wmma_f32_16x16x32_f16(false, a1, false, b1, (short)0, acc, false, false);
#pragma unroll
        for (int vv = 0; vv < 8; ++vv)               // D layout: M = vv + 8h, N = l15
          sG[s][t * 16 + h * 8 + vv][ncol] = (_Float16)acc[vv];
      }
    }
  }
  __syncthreads();

  // ---- Stage 2: spin-selected contraction  A_ij = +/- G_s[row]·Phi[row] ----
  for (int it = 0; it < PAIRS / 32; ++it) {              // 9 iterations, all lanes
    const int idx = it * 32 + lane;                      // = b*36 + q
    const int b = idx / 36, q = idx % 36;
    const int i = q / 6,   j = q % 6;
    const int si = sSpin[b * 6 + i], sj = sSpin[b * 6 + j];
    int sSel, rG, rP; float sgn = 1.0f;
    if (si == 0 && sj == 0)      { sSel = 0; rG = b * 6 + i; rP = b * 6 + j; }
    else if (si == 1 && sj == 1) { sSel = 1; rG = b * 6 + i; rP = b * 6 + j; }
    else if (si == 0)            { sSel = 2; rG = b * 6 + i; rP = b * 6 + j; }          // up,dn
    else                         { sSel = 2; rG = b * 6 + j; rP = b * 6 + i; sgn = -1.0f; } // dn,up: -A_ud^T
    float acc = 0.0f;
#pragma unroll
    for (int k8 = 0; k8 < 8; ++k8) {
      half8 gv = *(const half8*)&sG[sSel][rG][k8 * 8];
      half8 pv = *(const half8*)&sPhi[rP][k8 * 8];
#pragma unroll
      for (int uu = 0; uu < 8; ++uu) acc += (float)gv[uu] * (float)pv[uu];
    }
    sAp[idx] = sgn * acc;
  }
  __syncthreads();

  // ---- Stage 3: pair-correction MLP; hidden 32x32 layer on WMMA ----
  v16h w2f0, w2f1;
  {
    const _Float16* wb0 = &sW2t[l15][0];
    const _Float16* wb1 = &sW2t[16 + l15][0];
    w2f0 = cat8(*(const half8*)(wb0 + h * 8), *(const half8*)(wb0 + 16 + h * 8));
    w2f1 = cat8(*(const half8*)(wb1 + h * 8), *(const half8*)(wb1 + 16 + h * 8));
  }
  const float bn0 = sb2[l15],      bn1 = sb2[16 + l15];
  const float w3a = sW3[l15],      w3b = sW3[16 + l15];

  for (int mt = 0; mt < PT; ++mt) {
    const int p = mt * 16 + l15;                          // both halves: same pair
    const int b = p / 36, q = p % 36;
    const int i = q / 6,  j = q % 6;
    const float dx0 = sX[b * 6 + i][0] - sX[b * 6 + j][0];
    const float dx1 = sX[b * 6 + i][1] - sX[b * 6 + j][1];
    const float r2  = dx0 * dx0 + dx1 * dx1;
    const float sm  = (sSpin[b * 6 + i] == sSpin[b * 6 + j]) ? 1.0f : 0.0f;

    // layer 1: each lane computes only its half's 16 hidden units (A-frag layout)
    v16h ah;
#pragma unroll
    for (int e = 0; e < 8; ++e) {
      const int n0 = h * 8 + e;
      const int n1 = 16 + h * 8 + e;
      float z0 = dx0 * sW1[0][n0] + dx1 * sW1[1][n0] + r2 * sW1[2][n0] + sm * sW1[3][n0] + sb1[n0];
      float z1 = dx0 * sW1[0][n1] + dx1 * sW1[1][n1] + r2 * sW1[2][n1] + sm * sW1[3][n1] + sb1[n1];
      ah[e]     = (_Float16)silu_f(z0);
      ah[e + 8] = (_Float16)silu_f(z1);
    }
    // layer 2: h2 = h1 @ W2 (K=32 exactly -> one WMMA per 16-col tile)
    v8f c0 = {0.f,0.f,0.f,0.f,0.f,0.f,0.f,0.f};
    v8f c1 = {0.f,0.f,0.f,0.f,0.f,0.f,0.f,0.f};
    c0 = __builtin_amdgcn_wmma_f32_16x16x32_f16(false, ah, false, w2f0, (short)0, c0, false, false);
    c1 = __builtin_amdgcn_wmma_f32_16x16x32_f16(false, ah, false, w2f1, (short)0, c1, false, false);

    // layer 3 + SiLU, then reduce over the 16 lanes of each half (n-partials)
    float cb[8];
#pragma unroll
    for (int vv = 0; vv < 8; ++vv)
      cb[vv] = silu_f(c0[vv] + bn0) * w3a + silu_f(c1[vv] + bn1) * w3b;
#pragma unroll
    for (int msk = 1; msk < 16; msk <<= 1)
#pragma unroll
      for (int vv = 0; vv < 8; ++vv)
        cb[vv] += __shfl_xor(cb[vv], msk, 32);
#pragma unroll
    for (int vv = 0; vv < 8; ++vv)                        // pair M = vv + 8h of this tile
      if (l15 == vv) sOut[mt * 16 + h * 8 + vv] = cb[vv] + b3v;
  }
  __syncthreads();

  // ---- Stage 4: assemble A, 15-term Pfaffian, outputs ----
  if (lane < BPW) {
    const int b = lane;
    float aU[6][6];
#pragma unroll
    for (int i = 0; i < 6; ++i)
#pragma unroll
      for (int j = i + 1; j < 6; ++j)
        aU[i][j] = sAp[b * 36 + i * 6 + j]
                 + sc * (sOut[b * 36 + i * 6 + j] - sOut[b * 36 + j * 6 + i]);

    const int PI[15][6] = {
      {0,1,2,3,4,5},{0,1,2,4,3,5},{0,1,2,5,3,4},
      {0,2,1,3,4,5},{0,2,1,4,3,5},{0,2,1,5,3,4},
      {0,3,1,2,4,5},{0,3,1,4,2,5},{0,3,1,5,2,4},
      {0,4,1,2,3,5},{0,4,1,3,2,5},{0,4,1,5,2,3},
      {0,5,1,2,3,4},{0,5,1,3,2,4},{0,5,1,4,2,3}};
    float pf = 0.0f;
#pragma unroll
    for (int m = 0; m < 15; ++m) {
      float tmv = aU[PI[m][0]][PI[m][1]] * aU[PI[m][2]][PI[m][3]] * aU[PI[m][4]][PI[m][5]];
      pf += (m & 1) ? -tmv : tmv;                         // signs alternate +,-,+,...
    }
    const int gb = blockIdx.x * BPW + b;
    outp[gb]      = (pf > 0.0f) ? 1.0f : ((pf < 0.0f) ? -1.0f : 1.0f);
    outp[nB + gb] = 0.5f * logf(pf * pf + 1e-60f);
  }
}

extern "C" void kernel_launch(void* const* d_in, const int* in_sizes, int n_in,
                              void* d_out, int out_size, void* d_ws, size_t ws_size,
                              hipStream_t stream) {
  const float* x   = (const float*)d_in[0];
  const int*   spn = (const int*)  d_in[1];
  const float* Fud = (const float*)d_in[2];
  const float* Fuu = (const float*)d_in[3];
  const float* Fdd = (const float*)d_in[4];
  const float* W1  = (const float*)d_in[5];
  const float* b1  = (const float*)d_in[6];
  const float* W2  = (const float*)d_in[7];
  const float* b2  = (const float*)d_in[8];
  const float* W3  = (const float*)d_in[9];
  const float* b3  = (const float*)d_in[10];
  const float* sc  = (const float*)d_in[11];
  float* out = (float*)d_out;

  const int nB     = in_sizes[0] / (NPART * 2);   // 131072
  const int blocks = nB / BPW;                    // 16384 single-wave blocks
  pfaffian6_fused<<<blocks, 32, 0, stream>>>(x, spn, Fud, Fuu, Fdd,
                                             W1, b1, W2, b2, W3, b3, sc,
                                             out, nB);
}
